// advectionMassBlock_62663572849368
// MI455X (gfx1250) — matrix operand order repaired and checked by reading, not compiled, not run on hardware
//
#include <hip/hip_runtime.h>
#include <hip/hip_bf16.h>

// MI455X / gfx1250, wave32. Conv3x3 as implicit GEMM on v_wmma_f32_16x16x32_f16.
// Memory-bound pipeline => all chain intermediates stored as f16 (they are
// consumed at f16 by the next WMMA anyway), halving HBM traffic on those passes.

typedef __attribute__((ext_vector_type(16))) _Float16 v16h;
typedef __attribute__((ext_vector_type(8)))  _Float16 v8h;
typedef __attribute__((ext_vector_type(8)))  float    v8f;

#define B_ 8
#define C_ 32
#define H_ 256
#define W_ 256
#define PLANE (H_*W_)
#define NTOT (B_*C_*H_*W_)

// conv output tile per workgroup: TY rows x TX cols, 8 waves (1 row each)
#define TY 8
#define TX 16
#define HALO_Y (TY+2)
#define HALO_X (TX+2)
#define LDSX 20          // padded halo row stride (32 halves innermost)

// MODE 0: in0[b,c,h,w]
// MODE 1: tvec[b] * in0[0,c,h,w]          (t * TimeEmbed, broadcast batch)
// MODE 2: in0[b,c,h,w] + in1[b,c,h,w]     (x + te, in1 is f16)
template<int MODE, typename TIN, typename TOUT>
__global__ __launch_bounds__(256)
void conv3x3_wmma(const TIN* __restrict__ in0,
                  const _Float16* __restrict__ in1,
                  const float* __restrict__ tvec,
                  const float* __restrict__ wgt,   // (O=32, I=32, 3, 3) f32
                  const float* __restrict__ bias,  // (32,)
                  TOUT* __restrict__ out)
{
    __shared__ _Float16 sIn[HALO_Y][LDSX][C_];   // halo input tile, f16
    __shared__ _Float16 sW[9*C_*C_];             // [o][tap*32 + i], f16
    __shared__ float    sOut[TY][C_][TX];        // D staging for coalesced store

    const int tid = threadIdx.x;
    const int x0 = blockIdx.x * TX;
    const int y0 = blockIdx.y * TY;
    const int b  = blockIdx.z;

    // ---- weights: transform OIHW -> [o][ (ky*3+kx)*32 + i ] (B-matrix friendly)
    for (int e = tid; e < 9*C_*C_; e += 256) {
        int o   = e / 288;
        int r   = e - o*288;
        int tap = r >> 5;
        int i   = r & 31;
        int ky = tap/3, kx = tap - ky*3;
        sW[e] = (_Float16)wgt[((o*C_ + i)*3 + ky)*3 + kx];
    }

    float tval = 0.f;
    if (MODE == 1) tval = tvec[b];

    // ---- input halo tile (zero-padded SAME) -> f16, x-contiguous loads
    for (int e = tid; e < C_*HALO_Y*HALO_X; e += 256) {
        int c  = e / (HALO_Y*HALO_X);
        int r  = e - c*(HALO_Y*HALO_X);
        int hy = r / HALO_X;
        int hx = r - hy*HALO_X;
        int gy = y0 + hy - 1;
        int gx = x0 + hx - 1;
        float v = 0.f;
        if (gy >= 0 && gy < H_ && gx >= 0 && gx < W_) {
            int gidx = ((b*C_ + c)*H_ + gy)*W_ + gx;   // (B,C,H,W)
            if (MODE == 0)      v = (float)in0[gidx];
            else if (MODE == 1) v = tval * (float)in0[(c*H_ + gy)*W_ + gx]; // (1,C,H,W)
            else                v = (float)in0[gidx] + (float)in1[gidx];
        }
        sIn[hy][hx][c] = (_Float16)v;
    }
    __syncthreads();

    const int wid  = tid >> 5;        // wave -> output row within tile
    const int lane = tid & 31;
    const int mx   = lane & 15;       // M = x position within tile
    const int hi   = lane >> 4;       // lane half
    const int cA0  = hi * 8;          // A halves 0-7 hold K = cA0..cA0+7
    const int kB0  = hi * 8;          // B halves hold K = kB0..kB0+7, kB0+16..kB0+23

    v8f acc0 = {};                    // N tile 0 (channels 0-15)
    v8f acc1 = {};                    // N tile 1 (channels 16-31)

    #pragma unroll
    for (int tap = 0; tap < 9; ++tap) {        // K chunks: tap-major, 32 ch each
        const int ky = tap / 3, kx = tap - ky*3;

        union { v16h v; v8h h[2]; } A;
        const _Float16* ap = &sIn[wid + ky][mx + kx][0];
        A.h[0] = *(const v8h*)(ap + cA0);            // K = cA0 .. cA0+7
        A.h[1] = *(const v8h*)(ap + cA0 + 16);       // K = cA0+16 .. cA0+23

        union { v16h v; v8h h[2]; } Bt0, Bt1;
        const _Float16* bp0 = &sW[((lane & 15)      )*288 + tap*32 + kB0];
        const _Float16* bp1 = &sW[((lane & 15) + 16 )*288 + tap*32 + kB0];
        Bt0.h[0] = *(const v8h*)(bp0);
        Bt0.h[1] = *(const v8h*)(bp0 + 16);
        Bt1.h[0] = *(const v8h*)(bp1);
        Bt1.h[1] = *(const v8h*)(bp1 + 16);

        acc0 = __builtin_amdgcn_wmma_f32_16x16x32_f16(
            false, A.v, false, Bt0.v, (short)0, acc0, false, false);
        acc1 = __builtin_amdgcn_wmma_f32_16x16x32_f16(
            false, A.v, false, Bt1.v, (short)0, acc1, false, false);
    }

    // ---- stage D (rows M = r + hi*8, col N = lane&15) then coalesced store
    #pragma unroll
    for (int r = 0; r < 8; ++r) {
        int m = r + hi*8;
        sOut[wid][(lane & 15)     ][m] = acc0[r];
        sOut[wid][(lane & 15) + 16][m] = acc1[r];
    }
    __syncthreads();

    for (int e = tid; e < TY*C_*TX; e += 256) {
        int w  = e / (C_*TX);
        int r  = e - w*(C_*TX);
        int n  = r / TX;
        int xi = r - n*TX;
        out[((b*C_ + n)*H_ + (y0 + w))*W_ + (x0 + xi)] =
            (TOUT)(sOut[w][n][xi] + bias[n]);
    }
}

// LayerNorm over (H,W) per (b,c) plane, fused with SiLU (all 4 LN sites in the
// reference are followed by SiLU). f16 in/out, f32 accumulation.
__global__ __launch_bounds__(256)
void ln_silu(const _Float16* __restrict__ in, const float* __restrict__ lw,
             const float* __restrict__ lb, _Float16* __restrict__ out)
{
    __shared__ float s1[256], s2[256];
    const _Float16* p = in  + (size_t)blockIdx.x * PLANE;
    _Float16*       q = out + (size_t)blockIdx.x * PLANE;

    float s = 0.f, ss = 0.f;
    for (int i = threadIdx.x; i < PLANE; i += 256) {
        float v = (float)p[i]; s += v; ss += v*v;
    }
    s1[threadIdx.x] = s; s2[threadIdx.x] = ss;
    __syncthreads();
    for (int o = 128; o > 0; o >>= 1) {
        if (threadIdx.x < o) {
            s1[threadIdx.x] += s1[threadIdx.x + o];
            s2[threadIdx.x] += s2[threadIdx.x + o];
        }
        __syncthreads();
    }
    const float mu  = s1[0] * (1.f/PLANE);
    const float var = s2[0] * (1.f/PLANE) - mu*mu;
    const float rs  = rsqrtf(var + 1e-5f);

    for (int i = threadIdx.x; i < PLANE; i += 256) {
        float y = ((float)p[i] - mu) * rs * lw[i] + lb[i];
        q[i] = (_Float16)(y * (1.f / (1.f + __expf(-y))));   // SiLU
    }
}

__global__ __launch_bounds__(256)
void fill_zero(float* __restrict__ p)
{
    p[(size_t)blockIdx.x * 256 + threadIdx.x] = 0.f;
}

// forward bilinear splat, weights exactly as in the reference
__global__ __launch_bounds__(256)
void advect_splat(const float* __restrict__ img, const float* __restrict__ U,
                  const float* __restrict__ V, float* __restrict__ out)
{
    size_t i  = (size_t)blockIdx.x * 256 + threadIdx.x;
    int hw    = (int)(i & (PLANE - 1));
    int y     = hw >> 8;                 // W_ == 256
    int x     = hw & (W_ - 1);
    size_t base = i - (size_t)hw;

    float u = U[i], v = V[i];
    float Xn = fminf(fmaxf((float)x + u, 0.f), (float)(W_-1));
    float Yn = fminf(fmaxf((float)y + v, 0.f), (float)(H_-1));
    int x0i = (int)floorf(Xn);
    int y0i = (int)floorf(Yn);
    int x1i = min(x0i + 1, W_-1);
    int y1i = min(y0i + 1, H_-1);
    float fx0 = Xn - (float)x0i, fy0 = Yn - (float)y0i;
    float fx1 = (float)x1i - Xn, fy1 = (float)y1i - Yn;
    float src = img[i];

    atomicAdd(&out[base + (size_t)(y0i*W_ + x0i)], src * fx0 * fy0);
    atomicAdd(&out[base + (size_t)(y1i*W_ + x0i)], src * fx0 * fy1);
    atomicAdd(&out[base + (size_t)(y0i*W_ + x1i)], src * fx1 * fy0);
    atomicAdd(&out[base + (size_t)(y1i*W_ + x1i)], src * fx1 * fy1);
}

extern "C" void kernel_launch(void* const* d_in, const int* in_sizes, int n_in,
                              void* d_out, int out_size, void* d_ws, size_t ws_size,
                              hipStream_t stream)
{
    const float* x    = (const float*)d_in[0];
    const float* t    = (const float*)d_in[1];
    const float* TEU  = (const float*)d_in[2];
    const float* TEV  = (const float*)d_in[3];
    const float* tnu_w1 = (const float*)d_in[4];
    const float* tnu_b1 = (const float*)d_in[5];
    const float* tnu_lnw= (const float*)d_in[6];
    const float* tnu_lnb= (const float*)d_in[7];
    const float* tnu_w2 = (const float*)d_in[8];
    const float* tnu_b2 = (const float*)d_in[9];
    const float* tnv_w1 = (const float*)d_in[10];
    const float* tnv_b1 = (const float*)d_in[11];
    const float* tnv_lnw= (const float*)d_in[12];
    const float* tnv_lnb= (const float*)d_in[13];
    const float* tnv_w2 = (const float*)d_in[14];
    const float* tnv_b2 = (const float*)d_in[15];
    const float* cU1_w  = (const float*)d_in[16];
    const float* cU1_b  = (const float*)d_in[17];
    const float* lnU_w  = (const float*)d_in[18];
    const float* lnU_b  = (const float*)d_in[19];
    const float* cU2_w  = (const float*)d_in[20];
    const float* cU2_b  = (const float*)d_in[21];
    const float* cV1_w  = (const float*)d_in[22];
    const float* cV1_b  = (const float*)d_in[23];
    const float* lnV_w  = (const float*)d_in[24];
    const float* lnV_b  = (const float*)d_in[25];
    const float* cV2_w  = (const float*)d_in[26];
    const float* cV2_b  = (const float*)d_in[27];
    float* outp = (float*)d_out;

    // workspace: 2 f32 tensors (U, V) then 3 f16 tensors (chain intermediates)
    float*    Uf = (float*)d_ws;
    float*    Vf = Uf + (size_t)NTOT;
    _Float16* h0 = (_Float16*)(Vf + (size_t)NTOT);
    _Float16* h1 = h0 + (size_t)NTOT;
    _Float16* h2 = h1 + (size_t)NTOT;

    dim3 cg(W_/TX, H_/TY, B_);           // 16 x 32 x 8 workgroups
    dim3 cb(256);
    dim3 lg(B_*C_), lb(256);
    dim3 eg(NTOT/256), eb(256);

    // ---- U branch: teU = conv(LNSiLU(conv(t*TEU))), U = conv(LNSiLU(conv(x+teU)))
    conv3x3_wmma<1, float,    _Float16><<<cg, cb, 0, stream>>>(TEU, nullptr, t, tnu_w1, tnu_b1, h0);
    ln_silu<<<lg, lb, 0, stream>>>(h0, tnu_lnw, tnu_lnb, h1);
    conv3x3_wmma<0, _Float16, _Float16><<<cg, cb, 0, stream>>>(h1, nullptr, t, tnu_w2, tnu_b2, h0); // teU
    conv3x3_wmma<2, float,    _Float16><<<cg, cb, 0, stream>>>(x, h0, t, cU1_w, cU1_b, h1);
    ln_silu<<<lg, lb, 0, stream>>>(h1, lnU_w, lnU_b, h2);
    conv3x3_wmma<0, _Float16, float   ><<<cg, cb, 0, stream>>>(h2, nullptr, t, cU2_w, cU2_b, Uf);   // U

    // ---- V branch
    conv3x3_wmma<1, float,    _Float16><<<cg, cb, 0, stream>>>(TEV, nullptr, t, tnv_w1, tnv_b1, h0);
    ln_silu<<<lg, lb, 0, stream>>>(h0, tnv_lnw, tnv_lnb, h1);
    conv3x3_wmma<0, _Float16, _Float16><<<cg, cb, 0, stream>>>(h1, nullptr, t, tnv_w2, tnv_b2, h0); // teV
    conv3x3_wmma<2, float,    _Float16><<<cg, cb, 0, stream>>>(x, h0, t, cV1_w, cV1_b, h1);
    ln_silu<<<lg, lb, 0, stream>>>(h1, lnV_w, lnV_b, h2);
    conv3x3_wmma<0, _Float16, float   ><<<cg, cb, 0, stream>>>(h2, nullptr, t, cV2_w, cV2_b, Vf);   // V

    // ---- advect: zero output, then 4-way bilinear atomic splat
    fill_zero<<<eg, eb, 0, stream>>>(outp);
    advect_splat<<<eg, eb, 0, stream>>>(x, Uf, Vf, outp);
}
